// GraphSAGETrafficPredictor_49864570306628
// MI455X (gfx1250) — matrix operand order repaired and checked by reading, not compiled, not run on hardware
//
#include <hip/hip_runtime.h>
#include <hip/hip_bf16.h>
#include <cstdint>

typedef __attribute__((ext_vector_type(16))) _Float16 v16h;
typedef __attribute__((ext_vector_type(8)))  float    v8f;
typedef unsigned int v4u __attribute__((ext_vector_type(4)));
typedef int          v4i __attribute__((ext_vector_type(4)));
typedef int          v8i __attribute__((ext_vector_type(8)));

#define NNODES 10000
#define NBATCH 8
#define MTOT   (NNODES * NBATCH)   // 80000 rows
#define NEDGES 160000
#define HID    128
#define FLATK  24                  // real K of layer 0 (padded to 32)
#define BN_EPS 1e-5f

// ---------------------------------------------------------------- utilities

__global__ void k_zero(float* __restrict__ p, int n) {
  int i = blockIdx.x * blockDim.x + threadIdx.x;
  if (i < n) p[i] = 0.0f;
}

__global__ void k_deg(const long long* __restrict__ dst, float* __restrict__ deg) {
  int e = blockIdx.x * blockDim.x + threadIdx.x;
  if (e < NEDGES) atomicAdd(&deg[(int)dst[e]], 1.0f);
}

__global__ void k_invdeg(const float* __restrict__ deg, float* __restrict__ invd) {
  int i = blockIdx.x * blockDim.x + threadIdx.x;
  if (i < NNODES) invd[i] = 1.0f / fmaxf(deg[i], 1.0f);
}

// h0[r*32 + k] = x[b][t][n][c] with r=b*N+n, k=t*2+c (k<24); K padded to 32 with 0
__global__ void k_build_h0(const float* __restrict__ x, float* __restrict__ h0) {
  int i = blockIdx.x * blockDim.x + threadIdx.x;   // MTOT*32 threads (exact)
  int k = i & 31;
  int r = i >> 5;
  int b = r / NNODES;
  int n = r - b * NNODES;
  float v = 0.0f;
  if (k < FLATK) {
    int t = k >> 1, c = k & 1;
    v = x[(((size_t)b * 12 + t) * NNODES + n) * 2 + c];
  }
  h0[i] = v;
}

// Convert a (Ksrc x Nsrc) row-major f32 weight into f16 WMMA B fragments.
// Fragment (kt,nt) covers K=kt*32..+31, N=nt*16..+15; stored as 32 lanes x 16 halfs:
//   lane<16 -> K offsets 0..7,16..23, lane>=16 -> K offsets 8..15,24..31; column = lane&15.
__global__ void k_convw(const float* __restrict__ W, _Float16* __restrict__ frags,
                        int Ksrc, int Nsrc, int NT, int total) {
  int i = blockIdx.x * blockDim.x + threadIdx.x;
  if (i >= total) return;
  int e    = i & 15;
  int lane = (i >> 4) & 31;
  int frag = i >> 9;
  int kt = frag / NT, nt = frag - kt * NT;
  int K = kt * 32 + ((e & 8) ? 16 : 0) + ((lane & 16) ? 8 : 0) + (e & 7);
  int N = nt * 16 + (lane & 15);
  float v = (K < Ksrc && N < Nsrc) ? W[(size_t)K * Nsrc + N] : 0.0f;
  frags[i] = (_Float16)v;
}

// One wave per edge; 32 lanes cover K features (K/32 per lane), loop over batches.
__global__ void k_scatter(const float* __restrict__ h, float* agg,
                          const long long* __restrict__ src,
                          const long long* __restrict__ dst, int K) {
  int gid  = blockIdx.x * blockDim.x + threadIdx.x;
  int e    = gid >> 5;
  int lane = gid & 31;
  if (e >= NEDGES) return;
  int s = (int)src[e], d = (int)dst[e];
  int fpl = K >> 5;
  for (int b = 0; b < NBATCH; ++b) {
    const float* hp = h   + ((size_t)b * NNODES + s) * K + lane * fpl;
    float*       ap = agg + ((size_t)b * NNODES + d) * K + lane * fpl;
    if (b + 1 < NBATCH)
      __builtin_prefetch(hp + (size_t)NNODES * K, 0, 0);   // global_prefetch_b8
    #pragma unroll 4
    for (int f = 0; f < fpl; ++f) atomicAdd(ap + f, hp[f]);
  }
}

// ------------------------------------------------- TDM: global -> LDS staging

// Build a 1-D Tensor DMA Descriptor (D#) and issue tensor_load_to_lds.
// n8 = number of 8-byte elements to copy (data_size=3).
__device__ __forceinline__ void tdm_load_to_lds(unsigned lds_addr,
                                                const void* gptr, unsigned n8) {
  unsigned long long ga = (unsigned long long)gptr;
  v4u g0;
  g0[0] = 1u;                                             // count=1, user mode
  g0[1] = lds_addr;                                       // LDS byte address
  g0[2] = (unsigned)ga;                                   // global_addr[31:0]
  g0[3] = (unsigned)((ga >> 32) & 0x01FFFFFFu)            // global_addr[56:32]
        | (2u << 30);                                     // type=2 ("image")
  v8i g1;
  g1[0] = (int)(3u << 16);                                // data_size=3 (8B), mask=0
  g1[1] = (int)((n8 & 0xFFFFu) << 16);                    // tensor_dim0[15:0]
  g1[2] = (int)((n8 >> 16) | (1u << 16));                 // tensor_dim0[31:16] | tensor_dim1=1
  g1[3] = (int)((n8 & 0xFFFFu) << 16);                    // tile_dim0 = n8
  g1[4] = 1;                                              // tile_dim1 = 1
  g1[5] = (int)n8;                                        // tensor_dim0_stride[31:0]
  g1[6] = 0;
  g1[7] = 0;
  v4i z4 = {0, 0, 0, 0};
#if defined(__clang_major__) && (__clang_major__ >= 23)
  v8i z8 = {0, 0, 0, 0, 0, 0, 0, 0};
  __builtin_amdgcn_tensor_load_to_lds(g0, g1, z4, z4, z8, 0);  // clang-23 (6-arg)
#else
  __builtin_amdgcn_tensor_load_to_lds(g0, g1, z4, z4, 0);      // ROCm 7.2 (5-arg)
#endif
}

// ---------------------------------------------------------------- WMMA GEMMs

// Pack 8 floats (scaled) into halfs dst[base..base+7]; base is literal -> folds.
__device__ __forceinline__ void pack8(v16h& d, int base, const float4& a,
                                      const float4& b, float s) {
  d[base + 0] = (_Float16)(a.x * s); d[base + 1] = (_Float16)(a.y * s);
  d[base + 2] = (_Float16)(a.z * s); d[base + 3] = (_Float16)(a.w * s);
  d[base + 4] = (_Float16)(b.x * s); d[base + 5] = (_Float16)(b.y * s);
  d[base + 6] = (_Float16)(b.z * s); d[base + 7] = (_Float16)(b.w * s);
}

// Raw A-operand data for one 16x32 fragment (4x float4 per lane).
struct AF4 { float4 a0, a1, a2, a3; };

__device__ __forceinline__ AF4 load4(const float4* p, int idx) {
  AF4 r;
  r.a0 = p[idx];     r.a1 = p[idx + 1];   // k0 .. k0+7
  r.a2 = p[idx + 4]; r.a3 = p[idx + 5];   // k0+16 .. k0+23
  return r;
}

// A-fragment (16x32 f16, MxK): lane<16 holds row m=lane, K {k0..k0+7, k0+16..k0+23};
// lane>=16 holds row m=lane-16, K {k0+8..k0+15, k0+24..k0+31}.
__device__ __forceinline__ v16h mkfrag(const AF4& v, float s) {
  v16h a;
  pack8(a, 0, v.a0, v.a1, s);
  pack8(a, 8, v.a2, v.a3, s);
  return a;
}

// Fused SAGE layer: hout = ReLU(BN((agg*invdeg)@Wl + bl + h@Wr))
// Weights staged once per block into LDS by the Tensor Data Mover.
// A-loads software-pipelined one K-step ahead so global latency hides under WMMA.
// One wave owns 16 rows x all 128 columns -> hout may alias agg.
__global__ __launch_bounds__(256)
void k_sage_gemm(const float* h, const float* agg,
                 const float* __restrict__ invd,
                 const void* Wl, const void* Wr,
                 const float* __restrict__ bl,
                 const float* __restrict__ bn_g, const float* __restrict__ bn_b,
                 const float* __restrict__ bn_m, const float* __restrict__ bn_v,
                 float* hout, int K) {
  __shared__ _Float16 sWl[4 * 8 * 512];   // up to K=128: 32 frags * 512 halfs
  __shared__ _Float16 sWr[4 * 8 * 512];

  int lane = threadIdx.x & 31;
  int wave = threadIdx.x >> 5;
  int KT   = K >> 5;

  if (wave == 0) {
    unsigned n8 = (unsigned)(KT * 1024);  // 8-byte elements per weight matrix
    tdm_load_to_lds((unsigned)(unsigned long long)(void*)sWl, Wl, n8);
    tdm_load_to_lds((unsigned)(unsigned long long)(void*)sWr, Wr, n8);
    __builtin_amdgcn_s_wait_tensorcnt(0);
  }
  __syncthreads();

  int rowBase = (blockIdx.x * 8 + wave) * 16;
  int m  = lane & 15;
  int hi = lane >> 4;
  int row = rowBase + m;
  float idg = invd[row % NNODES];
  const float4* hp4 = (const float4*)(h   + (size_t)row * K);
  const float4* ap4 = (const float4*)(agg + (size_t)row * K);
  __builtin_prefetch(hp4, 0, 0);
  __builtin_prefetch(ap4, 0, 0);

  const v16h* sL = (const v16h*)sWl;
  const v16h* sR = (const v16h*)sWr;

  // prologue: issue kt=0 A-loads
  int idx = hi * 2;                       // float4 index of k-offset hi*8
  AF4 hv = load4(hp4, idx);
  AF4 gv = load4(ap4, idx);

  v8f acc[8] = {};
  for (int kt = 0; kt < KT; ++kt) {
    v16h ah = mkfrag(hv, 1.0f);
    v16h ag = mkfrag(gv, idg);
    if (kt + 1 < KT) {                    // pipeline next K-step's A-loads
      idx += 8;
      hv = load4(hp4, idx);
      gv = load4(ap4, idx);
    }
    // pipeline B fragments one N-tile ahead (LDS)
    v16h bl_f = sL[(kt * 8) * 32 + lane];
    v16h br_f = sR[(kt * 8) * 32 + lane];
    #pragma unroll
    for (int nt = 0; nt < 8; ++nt) {
      v16h cl = bl_f, cr = br_f;
      if (nt + 1 < 8) {
        bl_f = sL[(kt * 8 + nt + 1) * 32 + lane];
        br_f = sR[(kt * 8 + nt + 1) * 32 + lane];
      }
      acc[nt] = __builtin_amdgcn_wmma_f32_16x16x32_f16(false, ah, false, cr,
                                                       (short)0, acc[nt], false, false);
      acc[nt] = __builtin_amdgcn_wmma_f32_16x16x32_f16(false, ag, false, cl,
                                                       (short)0, acc[nt], false, false);
    }
  }

  // C layout: acc[nt][i], lane -> row = rowBase + i + 8*hi, col = nt*16 + (lane&15)
  #pragma unroll
  for (int nt = 0; nt < 8; ++nt) {
    int col = nt * 16 + m;
    float sc = bn_g[col] * rsqrtf(bn_v[col] + BN_EPS);
    float sh = bn_b[col] - bn_m[col] * sc;
    float bias = bl[col];
    #pragma unroll
    for (int i = 0; i < 8; ++i) {
      int r2 = rowBase + i + hi * 8;
      float v = (acc[nt][i] + bias) * sc + sh;
      hout[(size_t)r2 * HID + col] = fmaxf(v, 0.0f);
    }
  }
}

// Head: out(B,12,N,2) = h(80000,128) @ headW(128,24) + head_b, permuted store.
__global__ __launch_bounds__(256)
void k_head(const float* __restrict__ h, const v16h* __restrict__ Wf,
            const float* __restrict__ hb, float* __restrict__ out) {
  int lane = threadIdx.x & 31;
  int wave = threadIdx.x >> 5;
  int rowBase = (blockIdx.x * 8 + wave) * 16;
  int m  = lane & 15;
  int hi = lane >> 4;
  int row = rowBase + m;
  const float4* hp4 = (const float4*)(h + (size_t)row * HID);
  __builtin_prefetch(hp4, 0, 0);

  int idx = hi * 2;
  AF4 hv = load4(hp4, idx);
  v8f acc[2] = {};
  #pragma unroll
  for (int kt = 0; kt < 4; ++kt) {
    v16h ah = mkfrag(hv, 1.0f);
    if (kt + 1 < 4) {
      idx += 8;
      hv = load4(hp4, idx);
    }
    #pragma unroll
    for (int nt = 0; nt < 2; ++nt) {
      v16h bf = Wf[(kt * 2 + nt) * 32 + lane];
      acc[nt] = __builtin_amdgcn_wmma_f32_16x16x32_f16(false, ah, false, bf,
                                                       (short)0, acc[nt], false, false);
    }
  }
  #pragma unroll
  for (int nt = 0; nt < 2; ++nt) {
    int j = nt * 16 + m;
    if (j < 24) {
      float bias = hb[j];
      int t = j >> 1, c = j & 1;
      #pragma unroll
      for (int i = 0; i < 8; ++i) {
        int r2 = rowBase + i + hi * 8;
        int b = r2 / NNODES;
        int n = r2 - b * NNODES;
        out[(((size_t)b * 12 + t) * NNODES + n) * 2 + c] = acc[nt][i] + bias;
      }
    }
  }
}

// ---------------------------------------------------------------- launcher

extern "C" void kernel_launch(void* const* d_in, const int* in_sizes, int n_in,
                              void* d_out, int out_size, void* d_ws, size_t ws_size,
                              hipStream_t stream) {
  (void)in_sizes; (void)n_in; (void)out_size; (void)ws_size;
  const float*     x     = (const float*)d_in[0];
  const long long* ei    = (const long long*)d_in[1];
  const float*     Wl0   = (const float*)d_in[2];
  const float*     Wr0   = (const float*)d_in[3];
  const float*     bl0   = (const float*)d_in[4];
  const float*     Wl12  = (const float*)d_in[5];
  const float*     Wr12  = (const float*)d_in[6];
  const float*     bl12  = (const float*)d_in[7];
  const float*     bn_g  = (const float*)d_in[8];
  const float*     bn_b  = (const float*)d_in[9];
  const float*     bn_m  = (const float*)d_in[10];
  const float*     bn_v  = (const float*)d_in[11];
  const float*     headW = (const float*)d_in[12];
  const float*     headb = (const float*)d_in[13];
  float* out = (float*)d_out;
  const long long* src = ei;
  const long long* dst = ei + NEDGES;

  char* ws = (char*)d_ws;
  size_t off = 0;
  auto alloc = [&](size_t bytes) -> void* {
    void* p = ws + off;
    off = (off + bytes + 255) & ~(size_t)255;
    return p;
  };
  float* buf0 = (float*)alloc((size_t)MTOT * HID * sizeof(float));
  float* buf1 = (float*)alloc((size_t)MTOT * HID * sizeof(float));
  float* deg  = (float*)alloc((size_t)NNODES * sizeof(float));
  float* invd = (float*)alloc((size_t)NNODES * sizeof(float));
  _Float16* fWl0 = (_Float16*)alloc((size_t)1 * 8 * 512 * 2);
  _Float16* fWr0 = (_Float16*)alloc((size_t)1 * 8 * 512 * 2);
  _Float16* fWl1 = (_Float16*)alloc((size_t)4 * 8 * 512 * 2);
  _Float16* fWr1 = (_Float16*)alloc((size_t)4 * 8 * 512 * 2);
  _Float16* fWl2 = (_Float16*)alloc((size_t)4 * 8 * 512 * 2);
  _Float16* fWr2 = (_Float16*)alloc((size_t)4 * 8 * 512 * 2);
  _Float16* fHd  = (_Float16*)alloc((size_t)4 * 2 * 512 * 2);
  float* h0   = buf0;                       // MTOT x 32 (K-padded layer-0 input)
  float* agg0 = buf0 + (size_t)MTOT * 32;   // MTOT x 32 aggregation

  // degree + inverse degree
  k_zero  <<<(NNODES + 255) / 256, 256, 0, stream>>>(deg, NNODES);
  k_deg   <<<(NEDGES + 255) / 256, 256, 0, stream>>>(dst, deg);
  k_invdeg<<<(NNODES + 255) / 256, 256, 0, stream>>>(deg, invd);

  // input transpose (B,T,N,C) -> (B*N, 32 padded)
  k_build_h0<<<MTOT * 32 / 256, 256, 0, stream>>>(x, h0);

  // weights -> f16 WMMA fragments
  k_convw<<<16, 256, 0, stream>>>(Wl0,               fWl0,  24, 128, 8,  4096);
  k_convw<<<16, 256, 0, stream>>>(Wr0,               fWr0,  24, 128, 8,  4096);
  k_convw<<<64, 256, 0, stream>>>(Wl12,              fWl1, 128, 128, 8, 16384);
  k_convw<<<64, 256, 0, stream>>>(Wr12,              fWr1, 128, 128, 8, 16384);
  k_convw<<<64, 256, 0, stream>>>(Wl12 + 128 * 128,  fWl2, 128, 128, 8, 16384);
  k_convw<<<64, 256, 0, stream>>>(Wr12 + 128 * 128,  fWr2, 128, 128, 8, 16384);
  k_convw<<<16, 256, 0, stream>>>(headW,             fHd,  128,  24, 2,  4096);

  // ---- layer 0 (K=32 padded): buf0 -> buf1
  k_zero   <<<MTOT * 32 / 256, 256, 0, stream>>>(agg0, MTOT * 32);
  k_scatter<<<NEDGES * 32 / 256, 256, 0, stream>>>(h0, agg0, src, dst, 32);
  k_sage_gemm<<<MTOT / 128, 256, 0, stream>>>(h0, agg0, invd, fWl0, fWr0,
      bl0, bn_g, bn_b, bn_m, bn_v, buf1, 32);

  // ---- layer 1 (K=128): in buf1, agg buf0, out buf0 (row-exclusive alias OK)
  k_zero   <<<MTOT * 128 / 256, 256, 0, stream>>>(buf0, MTOT * 128);
  k_scatter<<<NEDGES * 32 / 256, 256, 0, stream>>>(buf1, buf0, src, dst, 128);
  k_sage_gemm<<<MTOT / 128, 256, 0, stream>>>(buf1, buf0, invd, fWl1, fWr1,
      bl12, bn_g + 128, bn_b + 128, bn_m + 128, bn_v + 128, buf0, 128);

  // ---- layer 2 (K=128): in buf0, agg buf1, out buf1
  k_zero   <<<MTOT * 128 / 256, 256, 0, stream>>>(buf1, MTOT * 128);
  k_scatter<<<NEDGES * 32 / 256, 256, 0, stream>>>(buf0, buf1, src, dst, 128);
  k_sage_gemm<<<MTOT / 128, 256, 0, stream>>>(buf0, buf1, invd, fWl2, fWr2,
      bl12 + 128, bn_g + 256, bn_b + 256, bn_m + 256, bn_v + 256, buf1, 128);

  // ---- head: buf1 @ headW -> out (permuted)
  k_head<<<MTOT / 128, 256, 0, stream>>>(buf1, (const v16h*)fHd, headb, out);
}
